// ConvVAE2d_10531259809828
// MI455X (gfx1250) — matrix operand order repaired and checked by reading, not compiled, not run on hardware
//
#include <hip/hip_runtime.h>
#include <hip/hip_bf16.h>

// ---------------------------------------------------------------------------
// ConvVAE2d forward on gfx1250: 4 GEMMs via v_wmma_f32_16x16x32_bf16.
// B=256, C=3, H=W=128, K=16 -> N_rows=16384, P=768, HID=2048, L=256.
// Block tile 128x256, 8 waves (2x4), wave tile 64x64, K-step 32,
// double-buffered LDS (1 barrier / K-step), L2 prefetch of k+2 tiles.
// LDS row stride padded to 40 elems (80B) -> conflict-free b128 frag reads.
// ---------------------------------------------------------------------------

typedef __attribute__((ext_vector_type(16))) __bf16 v16bf;
typedef __attribute__((ext_vector_type(8)))  float  v8f;

#define LDS_STRIDE 40   // elements per tile row (32 data + 8 pad) = 5 x uint4

union Frag {
    uint4 q[2];
    v16bf v;
};

__device__ __forceinline__ unsigned short f2bf(float f) {
    unsigned int u = __builtin_bit_cast(unsigned int, f);
    unsigned int r = u + 0x7FFFu + ((u >> 16) & 1u);   // round-to-nearest-even
    return (unsigned short)(r >> 16);
}

// --------------------------- prep kernels ----------------------------------

// patches[n, p] = x[b, c, gh*16+ky, gw*16+kx]  (bf16), n=(b,gh,gw), p=(c,ky,kx)
__global__ __launch_bounds__(256) void patch_extract_kernel(
    const float* __restrict__ x, unsigned short* __restrict__ patches, int total)
{
    int idx = blockIdx.x * 256 + threadIdx.x;
    if (idx >= total) return;
    int n = idx / 768;
    int p = idx - n * 768;
    int b  = n >> 6, s = n & 63, gh = s >> 3, gw = s & 7;
    int c  = p >> 8, rem = p & 255, ky = rem >> 4, kx = rem & 15;
    float v = x[((b * 3 + c) << 14) + ((gh * 16 + ky) << 7) + (gw * 16 + kx)];
    patches[idx] = f2bf(v);
}

// wt[n, k] = bf16(w[k, n]) : w is row-major [Kd, Nd]
__global__ __launch_bounds__(256) void transpose_bf16_kernel(
    const float* __restrict__ w, unsigned short* __restrict__ wt, int Kd, int Nd)
{
    int idx = blockIdx.x * 256 + threadIdx.x;
    if (idx >= Kd * Nd) return;
    int n = idx / Kd;
    int k = idx - n * Kd;
    wt[idx] = f2bf(w[k * Nd + n]);
}

// --------------------------- WMMA GEMM -------------------------------------
// C[M,N] = act(A[M,K] @ B[K,N] + bias), A bf16 row-major, B given as Bt[N,K].
// mode 0: out_bf[row*N+col] (he / hd)
// mode 1: out_f32 = mu_logvar in (B,512,8,8) layout; out_bf = z (cols<256, ld=256)
// mode 2: out_f32 = recon in (B,3,128,128) layout (cols decode to c,ky,kx)
__global__ __launch_bounds__(256, 1) void gemm_bf16_wmma_kernel(
    const unsigned short* __restrict__ A,
    const unsigned short* __restrict__ Bt,
    const float* __restrict__ bias,
    int M, int N, int Kt,
    int act, int mode,
    unsigned short* __restrict__ out_bf,
    float* __restrict__ out_f32)
{
    // double-buffered padded tiles: A 128x40 (10KB), B 256x40 (20KB) per buffer
    __shared__ __align__(16) unsigned short As[2][128 * LDS_STRIDE];
    __shared__ __align__(16) unsigned short Bs[2][256 * LDS_STRIDE];

    const int bm   = blockIdx.y * 128;
    const int bn   = blockIdx.x * 256;
    const int tid  = threadIdx.x;
    const int wave = tid >> 5;
    const int lane = tid & 31;
    const int wm   = (wave >> 2) * 64;   // 0,64
    const int wn   = (wave & 3) * 64;    // 0,64,128,192
    const int h    = lane >> 4;          // lane half
    const int lm   = lane & 15;

    v8f acc[4][4];
#pragma unroll
    for (int i = 0; i < 4; ++i)
#pragma unroll
        for (int j = 0; j < 4; ++j)
            acc[i][j] = (v8f){0.f, 0.f, 0.f, 0.f, 0.f, 0.f, 0.f, 0.f};

    const int nk = Kt >> 5;

    // ---- preload tile 0 into buffer 0 ----
#pragma unroll
    for (int p = 0; p < 2; ++p) {
        int e = p * 256 + tid, row = e >> 2, seg = e & 3;
        *((uint4*)&As[0][row * LDS_STRIDE] + seg) =
            *((const uint4*)(A + (bm + row) * Kt) + seg);
    }
#pragma unroll
    for (int p = 0; p < 4; ++p) {
        int e = p * 256 + tid, row = e >> 2, seg = e & 3;
        *((uint4*)&Bs[0][row * LDS_STRIDE] + seg) =
            *((const uint4*)(Bt + (bn + row) * Kt) + seg);
    }
    __syncthreads();

    for (int kt = 0; kt < nk; ++kt) {
        const int cur = kt & 1, nxt = cur ^ 1;
        const int k1 = (kt + 1) << 5;

        // prefetch k+2 tiles into L2 (global_prefetch_b8)
        if (kt + 2 < nk) {
            int k2 = (kt + 2) << 5;
            __builtin_prefetch(A  + (bm + (tid >> 1)) * Kt + k2, 0, 1);
            __builtin_prefetch(Bt + (bn + (tid >> 1)) * Kt + k2, 0, 1);
        }

        // ---- compute from current buffer ----
        // keep all 4 B fragments resident, reload 1 A fragment per ms group
        Frag bfr[4];
#pragma unroll
        for (int ns = 0; ns < 4; ++ns) {
            const uint4* pb = (const uint4*)&Bs[cur][(wn + ns * 16 + lm) * LDS_STRIDE];
            bfr[ns].q[0] = pb[h * 2];      // K = 16h..16h+7
            bfr[ns].q[1] = pb[h * 2 + 1];  // K = 16h+8..16h+15
        }
#pragma unroll
        for (int ms = 0; ms < 4; ++ms) {
            Frag afr;
            const uint4* pa = (const uint4*)&As[cur][(wm + ms * 16 + lm) * LDS_STRIDE];
            afr.q[0] = pa[h * 2];          // K = 16h..16h+7
            afr.q[1] = pa[h * 2 + 1];      // K = 16h+8..16h+15
#pragma unroll
            for (int ns = 0; ns < 4; ++ns)
                acc[ms][ns] = __builtin_amdgcn_wmma_f32_16x16x32_bf16(
                    false, afr.v, false, bfr[ns].v,
                    (short)0, acc[ms][ns], false, false);
        }

        // ---- stage tile k+1 into the other buffer (short live ranges) ----
        if (kt + 1 < nk) {
#pragma unroll
            for (int p = 0; p < 2; ++p) {
                int e = p * 256 + tid, row = e >> 2, seg = e & 3;
                uint4 va = *((const uint4*)(A + (bm + row) * Kt + k1) + seg);
                *((uint4*)&As[nxt][row * LDS_STRIDE] + seg) = va;
            }
#pragma unroll
            for (int p = 0; p < 4; ++p) {
                int e = p * 256 + tid, row = e >> 2, seg = e & 3;
                uint4 vb = *((const uint4*)(Bt + (bn + row) * Kt + k1) + seg);
                *((uint4*)&Bs[nxt][row * LDS_STRIDE] + seg) = vb;
            }
        }
        __syncthreads();
    }

    // epilogue: D layout -> element (vgpr i, lane): M = i + 8h, N = lm
#pragma unroll
    for (int ns = 0; ns < 4; ++ns) {
        int col = bn + wn + ns * 16 + lm;
        float bv = bias[col];
#pragma unroll
        for (int ms = 0; ms < 4; ++ms) {
#pragma unroll
            for (int i = 0; i < 8; ++i) {
                int row = bm + wm + ms * 16 + h * 8 + i;
                float v = acc[ms][ns][i] + bv;
                if (act == 1)      v = fmaxf(v, 0.f);
                else if (act == 2) v = 1.f / (1.f + __expf(-v));

                if (mode == 0) {
                    out_bf[row * N + col] = f2bf(v);
                } else if (mode == 1) {
                    int b = row >> 6, s = row & 63;
                    out_f32[(b * 512 + col) * 64 + s] = v;        // (B,512,8,8)
                    if (col < 256) out_bf[(row << 8) + col] = f2bf(v);  // z
                } else {
                    int b = row >> 6, s = row & 63, gh = s >> 3, gw = s & 7;
                    int c = col >> 8, rem = col & 255, ky = rem >> 4, kx = rem & 15;
                    out_f32[b * 49152 + (c << 14) + ((gh * 16 + ky) << 7)
                            + (gw * 16 + kx)] = v;               // (B,3,128,128)
                }
            }
        }
    }
}

// --------------------------- launcher --------------------------------------

extern "C" void kernel_launch(void* const* d_in, const int* in_sizes, int n_in,
                              void* d_out, int out_size, void* d_ws, size_t ws_size,
                              hipStream_t stream) {
    const float* x      = (const float*)d_in[0];
    const float* w_enc1 = (const float*)d_in[1];
    const float* b_enc1 = (const float*)d_in[2];
    const float* w_enc2 = (const float*)d_in[3];
    const float* b_enc2 = (const float*)d_in[4];
    const float* w_dec1 = (const float*)d_in[5];
    const float* b_dec1 = (const float*)d_in[6];
    const float* w_dec2 = (const float*)d_in[7];
    const float* b_dec2 = (const float*)d_in[8];
    float* out = (float*)d_out;

    const int M = 16384;               // B*GH*GW
    char* ws = (char*)d_ws;
    size_t off = 0;
    unsigned short* patches = (unsigned short*)(ws + off); off += (size_t)M * 768 * 2;
    unsigned short* w1T     = (unsigned short*)(ws + off); off += (size_t)2048 * 768 * 2;
    unsigned short* w2T     = (unsigned short*)(ws + off); off += (size_t)512 * 2048 * 2;
    unsigned short* w3T     = (unsigned short*)(ws + off); off += (size_t)2048 * 256 * 2;
    unsigned short* w4T     = (unsigned short*)(ws + off); off += (size_t)768 * 2048 * 2;
    unsigned short* he      = (unsigned short*)(ws + off); off += (size_t)M * 2048 * 2;
    unsigned short* z       = (unsigned short*)(ws + off); off += (size_t)M * 256 * 2;
    unsigned short* hd      = (unsigned short*)(ws + off); off += (size_t)M * 2048 * 2;

    // prep: patches + transposed bf16 weights
    {
        int total = M * 768;
        patch_extract_kernel<<<(total + 255) / 256, 256, 0, stream>>>(x, patches, total);
    }
    transpose_bf16_kernel<<<(768 * 2048 + 255) / 256, 256, 0, stream>>>(w_enc1, w1T, 768, 2048);
    transpose_bf16_kernel<<<(2048 * 512 + 255) / 256, 256, 0, stream>>>(w_enc2, w2T, 2048, 512);
    transpose_bf16_kernel<<<(256 * 2048 + 255) / 256, 256, 0, stream>>>(w_dec1, w3T, 256, 2048);
    transpose_bf16_kernel<<<(2048 * 768 + 255) / 256, 256, 0, stream>>>(w_dec2, w4T, 2048, 768);

    float* mu_logvar_out = out + (size_t)256 * 3 * 128 * 128;   // after recon_x

    // L1: he = relu(patches @ w_enc1 + b1)          [16384,768]x[768,2048]
    gemm_bf16_wmma_kernel<<<dim3(2048 / 256, M / 128), 256, 0, stream>>>(
        patches, w1T, b_enc1, M, 2048, 768, /*act*/1, /*mode*/0, he, nullptr);
    // L2: mu_logvar = he @ w_enc2 + b2 (-> (B,512,8,8) + z)   [16384,2048]x[2048,512]
    gemm_bf16_wmma_kernel<<<dim3(512 / 256, M / 128), 256, 0, stream>>>(
        he, w2T, b_enc2, M, 512, 2048, /*act*/0, /*mode*/1, z, mu_logvar_out);
    // L3: hd = relu(z @ w_dec1 + b3)                [16384,256]x[256,2048]
    gemm_bf16_wmma_kernel<<<dim3(2048 / 256, M / 128), 256, 0, stream>>>(
        z, w3T, b_dec1, M, 2048, 256, /*act*/1, /*mode*/0, hd, nullptr);
    // L4: recon = sigmoid(hd @ w_dec2 + b4) -> (B,3,128,128)  [16384,2048]x[2048,768]
    gemm_bf16_wmma_kernel<<<dim3(768 / 256, M / 128), 256, 0, stream>>>(
        hd, w4T, b_dec2, M, 768, 2048, /*act*/2, /*mode*/2, nullptr, out);
}
